// MCGumbelTreeCell_31258771981071
// MI455X (gfx1250) — matrix operand order repaired and checked by reading, not compiled, not run on hardware
//
#include <hip/hip_runtime.h>
#include <math.h>

// Problem constants (from reference)
#define NN   16
#define SS   32
#define DD   256
#define BB   4
#define NB   64          // N*B
#define CHD  1024        // 4*D
#define EPSF 1e-20f

typedef _Float16 half_t;
typedef __attribute__((ext_vector_type(16))) _Float16 v16h;
typedef __attribute__((ext_vector_type(8)))  _Float16 v8h;
typedef __attribute__((ext_vector_type(8)))  float    v8f;

// ---------------------------------------------------------------------------
// WMMA fragment helpers (contiguous k order k = 16*(lane/16) + s, same for A and B, wave32)
// A 16x32 f16: lanes 0-15 hold row M=lane, K=0..15; lanes 16-31 hold
// row M=lane-16, K=16..31.  Two 16-byte loads + shuffle.
__device__ __forceinline__ v16h frag_a(const half_t* base, int stride, int lane, int kbase) {
  int row  = lane & 15;
  int koff = (lane >> 4) << 4;          // 0 or 16
  const half_t* p = base + row * stride + kbase + koff;
  v8h a0 = *(const v8h*)(p);
  v8h a1 = *(const v8h*)(p + 8);
  return __builtin_shufflevector(a0, a1, 0,1,2,3,4,5,6,7,8,9,10,11,12,13,14,15);
}
// B 32x16 f16 (weights pre-transposed to [N][K] row-major): lane holds column
// N=lane&15; lanes 0-15 K=0..15, lanes 16-31 K=16..31 -> one contiguous 32B load.
__device__ __forceinline__ v16h frag_b(const half_t* base, int stride, int lane, int kbase) {
  int col  = lane & 15;
  int koff = (lane >> 4) << 4;          // 0 or 16
  return *(const v16h*)(base + col * stride + kbase + koff);
}
__device__ __forceinline__ v8f wmma32(v16h a, v16h b, v8f c) {
  return __builtin_amdgcn_wmma_f32_16x16x32_f16(false, a, false, b, (short)0, c, false, false);
}
__device__ __forceinline__ float gelu_exact(float x) {
  return 0.5f * x * (1.0f + erff(x * 0.70710678118654752f));
}
__device__ __forceinline__ float sigm(float x) { return 1.0f / (1.0f + __expf(-x)); }

// Deterministic hash RNG (stand-in for threefry fold_in(1234, i))
__device__ __forceinline__ float rng01(unsigned a, unsigned b, unsigned c) {
  unsigned h = 1234u;
  h ^= a + 0x9E3779B9u + (h << 6) + (h >> 2);
  h ^= b + 0x9E3779B9u + (h << 6) + (h >> 2);
  h ^= c + 0x9E3779B9u + (h << 6) + (h >> 2);
  h ^= h >> 16; h *= 0x7FEB352Du; h ^= h >> 15; h *= 0x846CA68Bu; h ^= h >> 16;
  return ((h >> 8) + 0.5f) * (1.0f / 16777216.0f);
}

// ---------------------------------------------------------------------------
// One-time: f32 -> f16 transposed weight copies (B-fragment friendly), zero accu
__global__ void k_convert(const float* __restrict__ wordW, const float* __restrict__ w1,
                          const float* __restrict__ w2,
                          half_t* __restrict__ wordWt, half_t* __restrict__ w1t,
                          half_t* __restrict__ w2t, float* __restrict__ accu) {
  int idx = blockIdx.x * blockDim.x + threadIdx.x;
  int stride = gridDim.x * blockDim.x;
  for (int i = idx; i < 1024 * 1024; i += stride) {
    int n2 = i >> 10, k2 = i & 1023;
    w2t[i] = (half_t)w2[k2 * 1024 + n2];                       // w2t[n][k]
    if (i < 512 * 1024) { int n1 = i >> 9, k1 = i & 511;  w1t[i]    = (half_t)w1[k1 * 1024 + n1]; }
    if (i < 256 * 256)  { int n0 = i >> 8, k0 = i & 255;  wordWt[i] = (half_t)wordW[k0 * 256 + n0]; }
    if (i < NB) accu[i] = 0.0f;
  }
}

// ---------------------------------------------------------------------------
// state = LN(x @ word_W + b), replicated B times.  16-row tile per block.
__global__ void k_word(const float* __restrict__ x, const half_t* __restrict__ wWt,
                       const float* __restrict__ wb, const float* __restrict__ g,
                       const float* __restrict__ beta, float* __restrict__ stateA) {
  __shared__ __align__(16) half_t lx[16 * 264];   // x tile f16
  __shared__ float lc[16 * 260];                  // result tile f32
  const int tid = threadIdx.x, lane = tid & 31, wv = tid >> 5;
  const int m0 = blockIdx.x * 16;                 // global row = n*32+s
  for (int i = tid; i < 16 * 256; i += 256) {
    int rr = i >> 8, c = i & 255;
    lx[rr * 264 + c] = (half_t)x[(m0 + rr) * 256 + c];
  }
  __syncthreads();
  for (int nt = wv * 2; nt < wv * 2 + 2; ++nt) {  // 16 N-tiles over 8 waves
    v8f acc = {};
    #pragma unroll
    for (int ks = 0; ks < 8; ++ks) {
      v16h a = frag_a(lx, 264, lane, ks * 32);
      v16h b = frag_b(wWt + nt * (16 * 256), 256, lane, ks * 32);
      acc = wmma32(a, b, acc);
    }
    int nbase = nt * 16 + (lane & 15);
    int mbase = (lane >> 4) << 3;
    #pragma unroll
    for (int r = 0; r < 8; ++r) lc[(mbase + r) * 260 + nbase] = acc[r] + wb[nbase];
  }
  __syncthreads();
  // LayerNorm: 16 threads per row, shfl width-16 reduction
  int row = tid >> 4, sub = tid & 15;
  float v[16], s = 0.f, s2 = 0.f;
  #pragma unroll
  for (int j = 0; j < 16; ++j) { v[j] = lc[row * 260 + sub * 16 + j]; s += v[j]; s2 += v[j] * v[j]; }
  #pragma unroll
  for (int off = 8; off; off >>= 1) { s += __shfl_xor(s, off, 16); s2 += __shfl_xor(s2, off, 16); }
  float mean = s * (1.f / 256.f);
  float var  = s2 * (1.f / 256.f) - mean * mean;
  float inv  = rsqrtf(var + 1e-5f);
  int m = m0 + row, n = m >> 5, sidx = m & 31;
  #pragma unroll
  for (int j = 0; j < 16; ++j) {
    int c = sub * 16 + j;
    float o = (v[j] - mean) * inv * g[c] + beta[c];
    #pragma unroll
    for (int b = 0; b < BB; ++b)
      stateA[(((n * BB + b) * SS) + sidx) * DD + c] = o;
  }
}

// ---------------------------------------------------------------------------
// Tree cell: new = LN(f1*l + f2*r + ig*parent) with two WMMA GEMMs.
// Block = (nb, 16-position tile).  8 waves.  Gate-aligned N-tiling on GEMM2.
__global__ void k_cell(const float* __restrict__ state, int Sc,
                       const half_t* __restrict__ w1t, const half_t* __restrict__ w2t,
                       const float* __restrict__ b1, const float* __restrict__ b2,
                       const float* __restrict__ g2, const float* __restrict__ bb2,
                       float* __restrict__ ns) {
  __shared__ __align__(16) half_t lcat[16 * 520];   // [l|r] tile f16
  __shared__ __align__(16) half_t li[16 * 1032];    // gelu(inter) tile f16
  __shared__ float sums[16], sums2[16];
  const int tid = threadIdx.x, lane = tid & 31, wv = tid >> 5;
  const int nb = blockIdx.x, p0 = blockIdx.y * 16;
  const float* st = state + nb * SS * DD;
  if (tid < 16) { sums[tid] = 0.f; sums2[tid] = 0.f; }
  for (int i = tid; i < 16 * 512; i += 256) {
    int rr = i >> 9, c = i & 511;
    int p = p0 + rr;
    float v = 0.f;
    if (p < Sc) v = (c < 256) ? st[p * DD + c] : st[(p + 1) * DD + (c - 256)];
    lcat[rr * 520 + c] = (half_t)v;
  }
  __syncthreads();
  // GEMM1: 16x512 @ 512x1024, K-steps of 32, 64 N-tiles over 8 waves; fuse GELU.
  for (int t = 0; t < 8; ++t) {
    int nt = wv * 8 + t;
    v8f acc = {};
    #pragma unroll
    for (int ks = 0; ks < 16; ++ks) {
      v16h a = frag_a(lcat, 520, lane, ks * 32);
      v16h b = frag_b(w1t + nt * (16 * 512), 512, lane, ks * 32);
      acc = wmma32(a, b, acc);
    }
    int nbase = nt * 16 + (lane & 15);
    int mbase = (lane >> 4) << 3;
    #pragma unroll
    for (int r = 0; r < 8; ++r)
      li[(mbase + r) * 1032 + nbase] = (half_t)gelu_exact(acc[r] + b1[nbase]);
  }
  __syncthreads();
  // GEMM2: gate-aligned tiles {ct, ct+16, ct+32, ct+48} so the sigmoid gating
  // fuses in registers (channel c gets contents[c], [256+c], [512+c], [768+c]).
  float keep[2][8];
  #pragma unroll
  for (int cti = 0; cti < 2; ++cti) {
    int ct = wv + cti * 8;
    v8f a0 = {}, a1 = {}, a2 = {}, a3 = {};
    #pragma unroll 8
    for (int ks = 0; ks < 32; ++ks) {
      v16h a  = frag_a(li, 1032, lane, ks * 32);
      v16h bA = frag_b(w2t + (ct +  0) * (16 * 1024), 1024, lane, ks * 32);
      v16h bB = frag_b(w2t + (ct + 16) * (16 * 1024), 1024, lane, ks * 32);
      v16h bC = frag_b(w2t + (ct + 32) * (16 * 1024), 1024, lane, ks * 32);
      v16h bD = frag_b(w2t + (ct + 48) * (16 * 1024), 1024, lane, ks * 32);
      if (ks + 1 < 32)
        __builtin_prefetch(w2t + (ct + 0) * (16 * 1024) + (ks + 1) * 32, 0, 3);
      a0 = wmma32(a, bA, a0); a1 = wmma32(a, bB, a1);
      a2 = wmma32(a, bC, a2); a3 = wmma32(a, bD, a3);
    }
    int c     = ct * 16 + (lane & 15);
    int mbase = (lane >> 4) << 3;
    #pragma unroll
    for (int r = 0; r < 8; ++r) {
      int m = mbase + r;
      float f1 = sigm(a0[r] + b2[c]);
      float f2 = sigm(a1[r] + b2[256 + c]);
      float ig = sigm(a2[r] + b2[512 + c]);
      float pr =       a3[r] + b2[768 + c];
      float lv = (float)lcat[m * 520 + c];
      float rv = (float)lcat[m * 520 + 256 + c];
      keep[cti][r] = f1 * lv + f2 * rv + ig * pr;
    }
  }
  // Row mean/var: shfl-reduce the 16 same-row lanes, then LDS float atomics.
  #pragma unroll
  for (int cti = 0; cti < 2; ++cti)
    #pragma unroll
    for (int r = 0; r < 8; ++r) {
      float v = keep[cti][r];
      float s = v, s2v = v * v;
      #pragma unroll
      for (int off = 8; off; off >>= 1) { s += __shfl_xor(s, off, 16); s2v += __shfl_xor(s2v, off, 16); }
      if ((lane & 15) == 0) {
        int m = ((lane >> 4) << 3) + r;
        atomicAdd(&sums[m], s);
        atomicAdd(&sums2[m], s2v);
      }
    }
  __syncthreads();
  #pragma unroll
  for (int cti = 0; cti < 2; ++cti) {
    int ct = wv + cti * 8;
    int c  = ct * 16 + (lane & 15);
    #pragma unroll
    for (int r = 0; r < 8; ++r) {
      int m = ((lane >> 4) << 3) + r;
      int p = p0 + m;
      if (p < Sc) {
        float mean = sums[m] * (1.f / 256.f);
        float var  = sums2[m] * (1.f / 256.f) - mean * mean;
        float inv  = rsqrtf(var + 1e-5f);
        ns[(nb * SS + p) * DD + c] = (keep[cti][r] - mean) * inv * g2[c] + bb2[c];
      }
    }
  }
}

// ---------------------------------------------------------------------------
// Gumbel select: decision fits exactly in one wave32 (Sc<=31 -> lane==position)
__global__ void k_select(const float* __restrict__ state, int Sc, int iter,
                         const float* __restrict__ ns, const float* __restrict__ imask,
                         const float* __restrict__ decW, const float* __restrict__ decb,
                         float* __restrict__ stout, float* __restrict__ accu) {
  __shared__ float ldw[5 * DD];
  __shared__ float ldots[31 * 5];
  __shared__ int   lk;
  __shared__ float ldone;
  const int tid = threadIdx.x, lane = tid & 31, wv = tid >> 5;
  const int nb = blockIdx.x, n = nb >> 2;
  const float* st = state + nb * SS * DD;
  const float* nh = ns + nb * SS * DD;
  for (int i = tid; i < 5 * DD; i += 256) ldw[i] = decW[i];
  __syncthreads();
  // dots[p][j] = mask[p] * (new_h[p] . W_j), wave-parallel dot products
  for (int idx = wv; idx < Sc * 5; idx += 8) {
    int p = idx / 5, j = idx % 5;
    float part = 0.f;
    for (int e = lane; e < DD; e += 32) part += nh[p * DD + e] * ldw[j * DD + e];
    #pragma unroll
    for (int off = 16; off; off >>= 1) part += __shfl_xor(part, off, 32);
    if (lane == 0) ldots[idx] = part * imask[n * SS + iter + 1 + p];
  }
  __syncthreads();
  if (wv == 0) {
    int p = lane;
    bool valid = p < Sc;
    float logit = 0.f;
    if (valid) {
      #pragma unroll
      for (int j = 0; j < 5; ++j) {
        int q = p - 2 + j;
        if (q >= 0 && q < Sc) logit += ldots[q * 5 + j];
      }
      logit = (logit + decb[0]) * rsqrtf(5.0f * 256.0f);
    }
    float mp = valid ? imask[n * SS + iter + 1 + p] : 0.f;
    float u  = rng01((unsigned)iter, (unsigned)nb, (unsigned)p);
    float gn = -logf(-logf(u + EPSF) + EPSF);
    float z1 = valid ? (logit + gn) : -3.0e38f;   // gumbel-perturbed
    float z2 = valid ? logit : -3.0e38f;          // plain
    float m1 = z1, m2 = z2;
    #pragma unroll
    for (int off = 16; off; off >>= 1) {
      m1 = fmaxf(m1, __shfl_xor(m1, off, 32));
      m2 = fmaxf(m2, __shfl_xor(m2, off, 32));
    }
    float e1 = valid ? __expf(z1 - m1) : 0.f;
    float e2 = valid ? __expf(z2 - m2) : 0.f;
    float s1 = e1, s2 = e2;
    #pragma unroll
    for (int off = 16; off; off >>= 1) { s1 += __shfl_xor(s1, off, 32); s2 += __shfl_xor(s2, off, 32); }
    float t1 = valid ? (e1 / s1 * mp + EPSF) : 0.f;
    float t2 = valid ? (e2 / s2 * mp + EPSF) : 0.f;
    float ts1 = t1, ts2 = t2;
    #pragma unroll
    for (int off = 16; off; off >>= 1) { ts1 += __shfl_xor(ts1, off, 32); ts2 += __shfl_xor(ts2, off, 32); }
    float ys1 = t1 / ts1;                // y_soft (with gumbel)
    float ys2 = t2 / ts2;                // masked_softmax(logits)
    // argmax(ys1), first-max tie-break like jnp.argmax
    float bv = valid ? ys1 : -1.f; int bi = valid ? p : 1000;
    #pragma unroll
    for (int off = 16; off; off >>= 1) {
      float ov = __shfl_xor(bv, off, 32);
      int   oi = __shfl_xor(bi, off, 32);
      if (ov > bv || (ov == bv && oi < bi)) { bv = ov; bi = oi; }
    }
    float score = __shfl(ys2, bi, 32);
    float done  = imask[n * SS + iter + 1];
    if (lane == 0) {
      float a0 = accu[nb];
      accu[nb] = done * (a0 + logf(score + EPSF)) + (1.f - done) * a0;
      lk = bi;
      ldone = done;
    }
  }
  __syncthreads();
  int k = lk;
  float done = ldone;
  for (int idx = tid; idx < Sc * DD; idx += 256) {
    int p = idx >> 8, c = idx & 255;
    float merged = (p == k) ? nh[p * DD + c]
                 : (p < k)  ? st[p * DD + c]
                            : st[(p + 1) * DD + c];
    stout[(nb * SS + p) * DD + c] = done * merged + (1.f - done) * st[p * DD + c];
  }
}

// Last iteration (Sc==1): no select, just the done-mask combine.
__global__ void k_lastcombine(const float* __restrict__ state, const float* __restrict__ ns,
                              const float* __restrict__ imask, float* __restrict__ stout) {
  int nb = blockIdx.x, c = threadIdx.x, n = nb >> 2;
  float done = imask[n * SS + SS - 1];
  stout[nb * SS * DD + c] = done * ns[nb * SS * DD + c] + (1.f - done) * state[nb * SS * DD + c];
}

// out[n] = sum_b softmax(accu[n,:])[b] * state[n*B+b, 0, :]
__global__ void k_output(const float* __restrict__ state, const float* __restrict__ accu,
                         float* __restrict__ out) {
  int n = blockIdx.x, d = threadIdx.x;
  float a0 = accu[n * 4 + 0], a1 = accu[n * 4 + 1], a2 = accu[n * 4 + 2], a3 = accu[n * 4 + 3];
  float m = fmaxf(fmaxf(a0, a1), fmaxf(a2, a3));
  float e0 = __expf(a0 - m), e1 = __expf(a1 - m), e2 = __expf(a2 - m), e3 = __expf(a3 - m);
  float s = e0 + e1 + e2 + e3;
  out[n * DD + d] = (e0 * state[((n * 4 + 0) * SS) * DD + d] +
                     e1 * state[((n * 4 + 1) * SS) * DD + d] +
                     e2 * state[((n * 4 + 2) * SS) * DD + d] +
                     e3 * state[((n * 4 + 3) * SS) * DD + d]) / s;
}

// ---------------------------------------------------------------------------
extern "C" void kernel_launch(void* const* d_in, const int* in_sizes, int n_in,
                              void* d_out, int out_size, void* d_ws, size_t ws_size,
                              hipStream_t stream) {
  (void)in_sizes; (void)n_in; (void)out_size; (void)ws_size;
  const float* x     = (const float*)d_in[0];
  const float* imask = (const float*)d_in[1];
  const float* wordW = (const float*)d_in[2];
  const float* wordb = (const float*)d_in[3];
  const float* lng   = (const float*)d_in[4];
  const float* lnb   = (const float*)d_in[5];
  const float* w1    = (const float*)d_in[6];
  const float* b1    = (const float*)d_in[7];
  const float* w2    = (const float*)d_in[8];
  const float* b2    = (const float*)d_in[9];
  const float* ln2g  = (const float*)d_in[10];
  const float* ln2b  = (const float*)d_in[11];
  const float* decW  = (const float*)d_in[12];
  const float* decb  = (const float*)d_in[13];
  float* out = (float*)d_out;

  char* ws = (char*)d_ws;
  size_t off = 0;
  auto carve = [&](size_t bytes) -> void* {
    void* p = ws + off;
    off = (off + bytes + 255) & ~(size_t)255;
    return p;
  };
  half_t* wordWt = (half_t*)carve(256 * 256 * sizeof(half_t));
  half_t* w1t    = (half_t*)carve(512 * 1024 * sizeof(half_t));
  half_t* w2t    = (half_t*)carve(1024 * 1024 * sizeof(half_t));
  float*  stA    = (float*)carve((size_t)NB * SS * DD * sizeof(float));
  float*  stB    = (float*)carve((size_t)NB * SS * DD * sizeof(float));
  float*  nst    = (float*)carve((size_t)NB * SS * DD * sizeof(float));
  float*  accu   = (float*)carve(NB * sizeof(float));

  k_convert<<<2048, 256, 0, stream>>>(wordW, w1, w2, wordWt, w1t, w2t, accu);
  k_word<<<32, 256, 0, stream>>>(x, wordWt, wordb, lng, lnb, stA);

  float* cur = stA;
  float* nxt = stB;
  for (int i = 0; i < SS - 1; ++i) {
    int Sc = SS - 1 - i;
    dim3 grid(NB, (Sc + 15) / 16);
    k_cell<<<grid, 256, 0, stream>>>(cur, Sc, w1t, w2t, b1, b2, ln2g, ln2b, nst);
    if (i < SS - 2) {
      k_select<<<NB, 256, 0, stream>>>(cur, Sc, i, nst, imask, decW, decb, nxt, accu);
    } else {
      k_lastcombine<<<NB, 256, 0, stream>>>(cur, nst, imask, nxt);
    }
    float* t = cur; cur = nxt; nxt = t;
  }
  k_output<<<NN, 256, 0, stream>>>(cur, accu, out);
}
